// ESA_SMILES_84782654423849
// MI455X (gfx1250) — matrix operand order repaired and checked by reading, not compiled, not run on hardware
//
#include <hip/hip_runtime.h>
#include <hip/hip_bf16.h>

typedef __attribute__((ext_vector_type(16))) __bf16 v16bf;
typedef __attribute__((ext_vector_type(8)))  float  v8f;

#define NBATCH   128
#define CDIM     512
#define TILE_M   64
#define LDAP     528          // padded LDS row stride (bf16 elems)
#define KT_TILES 16           // 512 / 32
#define NEG_INF  (-3.0e38f)

union FragB16 { v16bf v; uint4 q[2]; };

__device__ __forceinline__ unsigned short f32_to_bf16(float f) {
  unsigned int u = __float_as_uint(f);
  u += 0x7FFFu + ((u >> 16) & 1u);          // round-to-nearest-even
  return (unsigned short)(u >> 16);
}
__device__ __forceinline__ float bf16_to_f32(unsigned short h) {
  return __uint_as_float(((unsigned int)h) << 16);
}
__device__ __forceinline__ float wave_sum(float v) {
#pragma unroll
  for (int off = 16; off >= 1; off >>= 1) v += __shfl_xor(v, off, 32);
  return v;
}

// ---------------------------------------------------------------------------
// Prologue 1: offs[b] = lower_bound(data_batch, b) for b = 0..128 (sorted input)
// ---------------------------------------------------------------------------
__global__ void batch_offsets_kernel(const int* __restrict__ db, int T,
                                     int* __restrict__ offs) {
  int b = threadIdx.x;
  if (b <= NBATCH) {
    int lo = 0, hi = T;
    while (lo < hi) {
      int mid = (lo + hi) >> 1;
      if (db[mid] < b) lo = mid + 1; else hi = mid;
    }
    offs[b] = lo;
  }
}

// ---------------------------------------------------------------------------
// Prologue 2: pack W (row-major [N=512][K=512] f32) into bf16 WMMA B-fragment
// order. Tile (kt, nt) covers K in [kt*32,kt*32+32), N in [nt*16,nt*16+16).
// Within a tile, lane l owns 16 contiguous bf16 (32 bytes):
//   element i -> K = kt*32 + (i>>3)*16 + (l>>4)*8 + (i&7), N = nt*16 + (l&15)
// so each wave loads a whole B fragment with two coalesced b128 loads.
// ---------------------------------------------------------------------------
__global__ void pack_w_kernel(const float* __restrict__ W1,
                              const float* __restrict__ W2,
                              unsigned short* __restrict__ W1p,
                              unsigned short* __restrict__ W2p) {
  const float* W = (blockIdx.y == 0) ? W1 : W2;
  unsigned short* Wp = (blockIdx.y == 0) ? W1p : W2p;
  int tileId = blockIdx.x;                 // kt*32 + nt
  int kt = tileId >> 5, nt = tileId & 31;
  int e0 = threadIdx.x * 4;
#pragma unroll
  for (int u = 0; u < 4; ++u) {
    int e = e0 + u;                        // 0..511 within tile
    int l = e >> 4, i = e & 15;
    int K = kt * 32 + ((i >> 3) << 4) + ((l >> 4) << 3) + (i & 7);
    int N = (nt << 4) + (l & 15);
    Wp[(size_t)tileId * 512 + e] = f32_to_bf16(W[(size_t)N * CDIM + K]);
  }
}

// ---------------------------------------------------------------------------
// Main fused kernel: one workgroup per batch, 8 waves (wave32).
// LN -> GEMM1 (bf16 WMMA) -> GEMM2 (bf16 WMMA) -> online softmax accumulation.
// Wave w owns feature columns [w*64, w*64+64) => race-free (M,S,N) in LDS.
// B fragments are loaded once per (nt, kt) and applied to 4 row-slice
// accumulators; all 4 A fragments are loaded into distinct registers before
// the 4 WMMAs so LDS loads overlap matrix-pipe execution.
// ---------------------------------------------------------------------------
__global__ __launch_bounds__(256, 1) void esa_fused_kernel(
    const float* __restrict__ hs,
    const int* __restrict__ offs,
    const unsigned short* __restrict__ W1p,
    const unsigned short* __restrict__ W2p,
    const float* __restrict__ gamma,
    const float* __restrict__ beta,
    const float* __restrict__ b1,
    float* __restrict__ out) {
  extern __shared__ __align__(16) char smem_raw[];
  unsigned short* Abuf = (unsigned short*)smem_raw;          // [64][528] bf16
  unsigned short* Cbuf = Abuf + TILE_M * LDAP;               // [64][528] bf16
  float* Mrun = (float*)(Cbuf + TILE_M * LDAP);              // [512]
  float* Srun = Mrun + CDIM;                                 // [512]
  float* Nrun = Srun + CDIM;                                 // [512]

  const int b    = blockIdx.x;
  const int tid  = threadIdx.x;
  const int wave = tid >> 5;
  const int lane = tid & 31;
  const int lo16 = lane & 15;
  const int hi16 = lane >> 4;

  const int start = offs[b];
  const int cnt   = offs[b + 1] - start;

  for (int d = tid; d < CDIM; d += 256) {
    Mrun[d] = NEG_INF; Srun[d] = 0.f; Nrun[d] = 0.f;
  }
  __syncthreads();

  const int nt0 = wave * 4;                 // n-tiles owned by this wave

  for (int tb = 0; tb * TILE_M < cnt; ++tb) {
    // ---------- Phase 1: LayerNorm 64 rows -> bf16 Abuf ----------
#pragma unroll 1
    for (int i = 0; i < 8; ++i) {
      int row  = wave * 8 + i;
      int nidx = tb * TILE_M + row;         // node index within batch
      if (nidx < cnt) {
        const float* rp = hs + (size_t)(start + nidx) * CDIM;
        float4 x[4];
        float s = 0.f, ss = 0.f;
#pragma unroll
        for (int it = 0; it < 4; ++it) {
          x[it] = *(const float4*)(rp + (it * 32 + lane) * 4);
          s  += x[it].x + x[it].y + x[it].z + x[it].w;
          ss += x[it].x * x[it].x + x[it].y * x[it].y +
                x[it].z * x[it].z + x[it].w * x[it].w;
        }
        s = wave_sum(s); ss = wave_sum(ss);
        float mu  = s * (1.f / 512.f);
        float var = ss * (1.f / 512.f) - mu * mu;
        float rsd = rsqrtf(var + 1e-5f);
#pragma unroll
        for (int it = 0; it < 4; ++it) {
          int c = (it * 32 + lane) * 4;
          float4 g  = *(const float4*)(gamma + c);
          float4 be = *(const float4*)(beta + c);
          float y0 = (x[it].x - mu) * rsd * g.x + be.x;
          float y1 = (x[it].y - mu) * rsd * g.y + be.y;
          float y2 = (x[it].z - mu) * rsd * g.z + be.z;
          float y3 = (x[it].w - mu) * rsd * g.w + be.w;
          uint2 p;
          p.x = (unsigned)f32_to_bf16(y0) | ((unsigned)f32_to_bf16(y1) << 16);
          p.y = (unsigned)f32_to_bf16(y2) | ((unsigned)f32_to_bf16(y3) << 16);
          *(uint2*)(Abuf + row * LDAP + c) = p;
        }
      } else {
        uint2 z; z.x = 0u; z.y = 0u;
#pragma unroll
        for (int it = 0; it < 4; ++it)
          *(uint2*)(Abuf + row * LDAP + (it * 32 + lane) * 4) = z;
      }
    }
    __syncthreads();

    // ---------- Phase 2: GEMM1 cap = LN @ W1^T (no bias) -> bf16 Cbuf ----------
#pragma unroll 1
    for (int j = 0; j < 4; ++j) {
      int nt = nt0 + j;
      v8f acc[4];
#pragma unroll
      for (int rs = 0; rs < 4; ++rs)
        acc[rs] = (v8f){0.f, 0.f, 0.f, 0.f, 0.f, 0.f, 0.f, 0.f};
      const unsigned short* bp = W1p + (size_t)nt * 512 + lane * 16;
      const unsigned short* ap0 = Abuf + lo16 * LDAP + hi16 * 8;
#pragma unroll 2
      for (int kt = 0; kt < KT_TILES; ++kt) {
        FragB16 bfr, af[4];
        bfr.q[0] = *(const uint4*)(bp);
        bfr.q[1] = *(const uint4*)(bp + 8);
        if (kt + 1 < KT_TILES) __builtin_prefetch(bp + 32 * 512, 0, 3);
        bp += 32 * 512;
#pragma unroll
        for (int rs = 0; rs < 4; ++rs) {
          const unsigned short* ap = ap0 + rs * 16 * LDAP + kt * 32;
          af[rs].q[0] = *(const uint4*)(ap);        // K +0..7
          af[rs].q[1] = *(const uint4*)(ap + 16);   // K +16..23
        }
#pragma unroll
        for (int rs = 0; rs < 4; ++rs)
          acc[rs] = __builtin_amdgcn_wmma_f32_16x16x32_bf16(
              false, af[rs].v, false, bfr.v, (short)0, acc[rs], false, false);
      }
#pragma unroll
      for (int rs = 0; rs < 4; ++rs)
#pragma unroll
        for (int r = 0; r < 8; ++r) {
          int m = rs * 16 + r + 8 * hi16;
          Cbuf[m * LDAP + nt * 16 + lo16] = f32_to_bf16(acc[rs][r]);
        }
    }
    __syncthreads();

    // ---------- Phase 3: GEMM2 feat = cap @ W2^T ; online softmax ----------
#pragma unroll 1
    for (int j = 0; j < 4; ++j) {
      int nt = nt0 + j;
      v8f acc[4];
#pragma unroll
      for (int rs = 0; rs < 4; ++rs)
        acc[rs] = (v8f){0.f, 0.f, 0.f, 0.f, 0.f, 0.f, 0.f, 0.f};
      const unsigned short* bp = W2p + (size_t)nt * 512 + lane * 16;
      const unsigned short* ap0 = Cbuf + lo16 * LDAP + hi16 * 8;
#pragma unroll 2
      for (int kt = 0; kt < KT_TILES; ++kt) {
        FragB16 bfr, af[4];
        bfr.q[0] = *(const uint4*)(bp);
        bfr.q[1] = *(const uint4*)(bp + 8);
        if (kt + 1 < KT_TILES) __builtin_prefetch(bp + 32 * 512, 0, 3);
        bp += 32 * 512;
#pragma unroll
        for (int rs = 0; rs < 4; ++rs) {
          const unsigned short* ap = ap0 + rs * 16 * LDAP + kt * 32;
          af[rs].q[0] = *(const uint4*)(ap);
          af[rs].q[1] = *(const uint4*)(ap + 16);
        }
#pragma unroll
        for (int rs = 0; rs < 4; ++rs)
          acc[rs] = __builtin_amdgcn_wmma_f32_16x16x32_bf16(
              false, af[rs].v, false, bfr.v, (short)0, acc[rs], false, false);
      }
      // Online softmax update for column n (lanes l and l+16 mirror n).
      int n = nt * 16 + lo16;
#pragma unroll 1
      for (int rs = 0; rs < 4; ++rs) {
        float v[8];
        float tmax = NEG_INF;
#pragma unroll
        for (int r = 0; r < 8; ++r) {
          int m = rs * 16 + r + 8 * hi16;
          bool ok = (tb * TILE_M + m) < cnt;
          v[r] = ok ? acc[rs][r] : NEG_INF;
          tmax = fmaxf(tmax, v[r]);
        }
        tmax = fmaxf(tmax, __shfl_xor(tmax, 16, 32));
        float Mo = Mrun[n];
        float Mn = fmaxf(Mo, tmax);
        float s_add = 0.f, n_add = 0.f;
        if (Mn > NEG_INF) {
#pragma unroll
          for (int r = 0; r < 8; ++r) {
            int m = rs * 16 + r + 8 * hi16;
            float e = (v[r] > NEG_INF) ? __expf(v[r] - Mn) : 0.f;
            float c = bf16_to_f32(Cbuf[m * LDAP + n]);
            s_add += e;
            n_add += e * c;
          }
        }
        s_add += __shfl_xor(s_add, 16, 32);
        n_add += __shfl_xor(n_add, 16, 32);
        if (lane < 16 && Mn > NEG_INF) {
          float scale = (Mo > NEG_INF) ? __expf(Mo - Mn) : 0.f;
          Srun[n] = Srun[n] * scale + s_add;
          Nrun[n] = Nrun[n] * scale + n_add;
          Mrun[n] = Mn;
        }
      }
    }
    __syncthreads();
  }

  // Epilogue: out = Numer/S + b1 (b1/b2 folded out of the softmax — see analysis)
  for (int d = tid; d < CDIM; d += 256) {
    float S = Srun[d];
    out[(size_t)b * CDIM + d] = (S > 0.f) ? (Nrun[d] / S + b1[d]) : 0.f;
  }
}

// ---------------------------------------------------------------------------
extern "C" void kernel_launch(void* const* d_in, const int* in_sizes, int n_in,
                              void* d_out, int out_size, void* d_ws, size_t ws_size,
                              hipStream_t stream) {
  const float* hs    = (const float*)d_in[0];
  const int*   db    = (const int*)d_in[1];
  // d_in[2] = max_nodes (unused: padded rows are provably inert)
  const float* gamma = (const float*)d_in[3];
  const float* beta  = (const float*)d_in[4];
  const float* W1    = (const float*)d_in[5];
  const float* b1    = (const float*)d_in[6];
  const float* W2    = (const float*)d_in[7];
  // d_in[8] = b2 (unused: per-column shift is softmax-invariant)
  float* out = (float*)d_out;
  const int T = in_sizes[1];

  char* ws = (char*)d_ws;
  int* offs = (int*)ws;                                        // 129 ints
  unsigned short* W1p = (unsigned short*)(ws + 1024);          // 512 KB
  unsigned short* W2p = (unsigned short*)(ws + 1024 + 512 * 1024);

  batch_offsets_kernel<<<1, 256, 0, stream>>>(db, T, offs);
  pack_w_kernel<<<dim3(512, 2), 128, 0, stream>>>(W1, W2, W1p, W2p);

  size_t smem = (size_t)TILE_M * LDAP * sizeof(unsigned short) * 2 +
                3 * CDIM * sizeof(float);                      // 141,312 B
  (void)hipFuncSetAttribute((const void*)esa_fused_kernel,
                            hipFuncAttributeMaxDynamicSharedMemorySize,
                            (int)smem);
  esa_fused_kernel<<<NBATCH, 256, smem, stream>>>(hs, offs, W1p, W2p,
                                                  gamma, beta, b1, out);
}
